// PerformerAttention_90675349553511
// MI455X (gfx1250) — matrix operand order repaired and checked by reading, not compiled, not run on hardware
//
#include <hip/hip_runtime.h>
#include <hip/hip_bf16.h>

// Performer (FAVOR+) linear attention for MI455X / gfx1250.
// B=2, T=8192, H=8, E=64, R=256. All four GEMMs on v_wmma_f32_16x16x32_bf16.
// A-fragments and B-fragments each load as two aligned ds_load_b128 per lane.

typedef __attribute__((ext_vector_type(16))) __bf16   v16bf;
typedef __attribute__((ext_vector_type(8)))  __bf16   v8bf;
typedef __attribute__((ext_vector_type(4)))  __bf16   v4bf;
typedef __attribute__((ext_vector_type(8)))  float    v8f;
typedef __attribute__((ext_vector_type(8)))  unsigned v8u;
typedef __attribute__((ext_vector_type(4)))  unsigned v4u;
typedef __attribute__((ext_vector_type(2)))  unsigned v2u;

constexpr int Bc = 2, Tc = 8192, Hc = 8, Ec = 64, Rc = 256;
constexpr int TILE = 64;            // T-rows per block
constexpr int NT   = Tc / TILE;     // 128 tiles along T
constexpr int NTHREADS = 256;       // 8 wave32

__device__ __forceinline__ v4bf cvt4(const float4 f) {
  v4bf r;
  r[0] = (__bf16)f.x; r[1] = (__bf16)f.y; r[2] = (__bf16)f.z; r[3] = (__bf16)f.w;
  return r;
}

// ---------------- WMMA fragment loaders (CDNA5 wave32 layouts) ----------------
// A: 16x32 bf16, row-major LDS [M][K]. Lane m = lane&15, half = lane>>4.
// VGPRs 0-3: K = 8*half + 0..7 (16B contig); VGPRs 4-7: K = 16 + 8*half + 0..7.
__device__ __forceinline__ v16bf load_frag_a(const __bf16* s, int ld, int row0, int k0) {
  const int lane = threadIdx.x & 31;
  const int half = lane >> 4;
  const int m    = lane & 15;
  const __bf16* base = s + (row0 + m) * ld + k0 + (half << 3);
  const v4u lo = *(const v4u*)(base);        // ds_load_b128
  const v4u hi = *(const v4u*)(base + 16);   // ds_load_b128
  v8u t;
#pragma unroll
  for (int i = 0; i < 4; ++i) { t[i] = lo[i]; t[i + 4] = hi[i]; }
  return __builtin_bit_cast(v16bf, t);
}

// B: 32x16 bf16 from K-contiguous (transposed) LDS sT[N][K], row stride ldk.
// Lane n = lane&15; K = 8*half + 0..7, then 16 + 8*half + 0..7 -> two aligned b128 loads.
__device__ __forceinline__ v16bf load_frag_b_t(const __bf16* sT, int ldk, int k0, int n0) {
  const int lane = threadIdx.x & 31;
  const int half = lane >> 4;
  const int n    = lane & 15;
  const __bf16* base = sT + (n0 + n) * ldk + k0 + (half << 3);
  const v4u lo = *(const v4u*)(base);
  const v4u hi = *(const v4u*)(base + 16);
  v8u t;
#pragma unroll
  for (int i = 0; i < 4; ++i) { t[i] = lo[i]; t[i + 4] = hi[i]; }
  return __builtin_bit_cast(v16bf, t);
}

__device__ __forceinline__ v8f wmma_bf16(v16bf a, v16bf b, v8f c) {
  return __builtin_amdgcn_wmma_f32_16x16x32_bf16(false, a, false, b, (short)0, c,
                                                 false, false);
}

// ---------------- workspace zero ----------------
__global__ void zero_ws_kernel(float* __restrict__ p, int n) {
  int i = blockIdx.x * blockDim.x + threadIdx.x;
  if (i < n) p[i] = 0.0f;
}

// ---------------- Pass 1: kv[r][e] and ksum[r] per (b,h) ----------------
__global__ __launch_bounds__(NTHREADS)
void perf_pass1(const float* __restrict__ key, const float* __restrict__ value,
                const float* __restrict__ w, float* __restrict__ gkv,
                float* __restrict__ gksum) {
  extern __shared__ char smem[];
  __bf16* sWt = (__bf16*)smem;           // [Rc][Ec]   W transposed (K=e contiguous)
  __bf16* sK  = sWt + Rc * Ec;           // [TILE][Ec] K tile row-major
  __bf16* sVt = sK + TILE * Ec;          // [Ec][TILE] V transposed (K=t contiguous)
  __bf16* sFt = sVt + Ec * TILE;         // [Rc][TILE] k_feats transposed
  float*  sNorm = (float*)(sFt + Rc * TILE);  // [TILE]

  const int tid  = threadIdx.x;
  const int bid  = blockIdx.x;
  const int tile = bid % NT;
  const int bh   = bid / NT;
  const int b    = bh / Hc;
  const int h    = bh % Hc;
  const int t0   = tile * TILE;

  // stage W transposed: sWt[r][e] = W[e][r]   (float4 reads over r)
  for (int i = tid; i < (Ec * Rc) / 4; i += NTHREADS) {
    const int r4 = (i & 63) * 4, e = i >> 6;
    const float4 f = *(const float4*)&w[((size_t)bh * Ec + e) * Rc + r4];
    const v4bf bf = cvt4(f);
#pragma unroll
    for (int j = 0; j < 4; ++j) sWt[(r4 + j) * Ec + e] = bf[j];
  }
  // stage K row-major (b64 stores) and V transposed
  for (int i = tid; i < (TILE * Ec) / 4; i += NTHREADS) {
    const int e4 = (i & 15) * 4, t = i >> 4;
    const size_t g = (((size_t)b * Tc + t0 + t) * Hc + h) * Ec + e4;
    const v4bf kb = cvt4(*(const float4*)&key[g]);
    const v4bf vb = cvt4(*(const float4*)&value[g]);
    *(v2u*)(sK + t * Ec + e4) = __builtin_bit_cast(v2u, kb);
#pragma unroll
    for (int j = 0; j < 4; ++j) sVt[(e4 + j) * TILE + t] = vb[j];
  }
  // fp32 row norms of K (float4 reads)
  if (tid < TILE) {
    const float4* base = (const float4*)&key[(((size_t)b * Tc + t0 + tid) * Hc + h) * Ec];
    float acc = 0.0f;
#pragma unroll
    for (int e = 0; e < Ec / 4; ++e) {
      const float4 x = base[e];
      acc += x.x * x.x + x.y * x.y + x.z * x.z + x.w * x.w;
    }
    sNorm[tid] = acc;
  }
  // prefetch next tile rows (gfx1250 global_prefetch_b8)
  if (tile + 1 < NT && tid < TILE) {
    const size_t nb = (((size_t)b * Tc + t0 + TILE + tid) * Hc + h) * Ec;
    __builtin_prefetch(&key[nb], 0, 1);
    __builtin_prefetch(&value[nb], 0, 1);
  }
  __syncthreads();

  const int wave = tid >> 5;
  const int lane = tid & 31;
  const float inv_sqrt_r = 0.0625f;  // 1/sqrt(256)

  // proj = Ktile @ W -> feats (transposed into sFt).
  // Wave owns one mt (of 4) and 8 of 16 nt's; A fragments held in registers.
  {
    const int mt     = wave >> 1;
    const int ntbase = (wave & 1) * 8;
    const v16bf a0 = load_frag_a(sK, Ec, mt * 16, 0);
    const v16bf a1 = load_frag_a(sK, Ec, mt * 16, 32);
    const int tbase = mt * 16 + ((lane >> 4) << 3);
    const int ncol  = lane & 15;
#pragma unroll
    for (int nn = 0; nn < 8; ++nn) {
      const int nt = ntbase + nn;
      v8f c = {};
      c = wmma_bf16(a0, load_frag_b_t(sWt, Ec, 0,  nt * 16), c);
      c = wmma_bf16(a1, load_frag_b_t(sWt, Ec, 32, nt * 16), c);
      const int rcol = nt * 16 + ncol;
      v8bf fb;
#pragma unroll
      for (int v = 0; v < 8; ++v)
        fb[v] = (__bf16)(__expf(c[v] - 0.5f * sNorm[tbase + v]) * inv_sqrt_r);
      *(v4u*)(sFt + rcol * TILE + tbase) = __builtin_bit_cast(v4u, fb);  // ds_store_b128
    }
  }
  __syncthreads();

  // ksum[r] += sum_t feats[r][t]   (contiguous LDS row)
  {
    float acc = 0.0f;
#pragma unroll 8
    for (int t = 0; t < TILE; ++t) acc += (float)sFt[tid * TILE + t];
    atomicAdd(&gksum[(size_t)bh * Rc + tid], acc);
  }

  // kv = featsT @ V : [Rc x Ec]. Wave owns one nt (of 4) and 8 of 16 mt's;
  // B fragments held in registers, A = sFt row-major (two b128 loads each).
  {
    const int nt     = wave & 3;
    const int mtbase = (wave >> 2) * 8;
    const v16bf b0 = load_frag_b_t(sVt, TILE, 0,  nt * 16);
    const v16bf b1 = load_frag_b_t(sVt, TILE, 32, nt * 16);
    const int rowb = (lane >> 4) << 3;
    const int col  = nt * 16 + (lane & 15);
#pragma unroll
    for (int mm = 0; mm < 8; ++mm) {
      const int mt = mtbase + mm;
      v8f c = {};
      c = wmma_bf16(load_frag_a(sFt, TILE, mt * 16, 0),  b0, c);
      c = wmma_bf16(load_frag_a(sFt, TILE, mt * 16, 32), b1, c);
#pragma unroll
      for (int v = 0; v < 8; ++v) {
        const int r = mt * 16 + rowb + v;
        atomicAdd(&gkv[((size_t)bh * Rc + r) * Ec + col], c[v]);
      }
    }
  }
}

// ---------------- Pass 2: out = (q_feats @ kv) / (q_feats . ksum) ----------------
__global__ __launch_bounds__(NTHREADS)
void perf_pass2(const float* __restrict__ query, const float* __restrict__ w,
                const float* __restrict__ gkv, const float* __restrict__ gksum,
                float* __restrict__ out) {
  extern __shared__ char smem[];
  __bf16* sWt  = (__bf16*)smem;          // [Rc][Ec]
  __bf16* sQ   = sWt + Rc * Ec;          // [TILE][Ec]
  __bf16* sF   = sQ + TILE * Ec;         // [TILE][Rc]  q_feats row-major
  __bf16* sKVt = sF + TILE * Rc;         // [Ec][Rc]    kv transposed (K=r contiguous)
  float* sKsum = (float*)(sKVt + Ec * Rc);  // [Rc]
  float* sNorm = sKsum + Rc;                // [TILE]
  float* sD    = sNorm + TILE;              // [TILE]

  const int tid  = threadIdx.x;
  const int bid  = blockIdx.x;
  const int tile = bid % NT;
  const int bh   = bid / NT;
  const int b    = bh / Hc;
  const int h    = bh % Hc;
  const int t0   = tile * TILE;

  for (int i = tid; i < (Ec * Rc) / 4; i += NTHREADS) {
    const int r4 = (i & 63) * 4, e = i >> 6;
    const v4bf bf = cvt4(*(const float4*)&w[((size_t)bh * Ec + e) * Rc + r4]);
#pragma unroll
    for (int j = 0; j < 4; ++j) sWt[(r4 + j) * Ec + e] = bf[j];
  }
  for (int i = tid; i < (TILE * Ec) / 4; i += NTHREADS) {
    const int e4 = (i & 15) * 4, t = i >> 4;
    const v4bf qb = cvt4(*(const float4*)&query[(((size_t)b * Tc + t0 + t) * Hc + h) * Ec + e4]);
    *(v2u*)(sQ + t * Ec + e4) = __builtin_bit_cast(v2u, qb);
  }
  // kv transposed: sKVt[e][r] = gkv[r][e]  (float4 reads over e)
  for (int i = tid; i < (Rc * Ec) / 4; i += NTHREADS) {
    const int e4 = (i & 15) * 4, r = i >> 4;
    const v4bf kb = cvt4(*(const float4*)&gkv[((size_t)bh * Rc + r) * Ec + e4]);
#pragma unroll
    for (int j = 0; j < 4; ++j) sKVt[(e4 + j) * Rc + r] = kb[j];
  }
  sKsum[tid] = gksum[(size_t)bh * Rc + tid];
  if (tid < TILE) {
    const float4* base = (const float4*)&query[(((size_t)b * Tc + t0 + tid) * Hc + h) * Ec];
    float acc = 0.0f;
#pragma unroll
    for (int e = 0; e < Ec / 4; ++e) {
      const float4 x = base[e];
      acc += x.x * x.x + x.y * x.y + x.z * x.z + x.w * x.w;
    }
    sNorm[tid] = acc;
  }
  __syncthreads();

  const int wave = tid >> 5;
  const int lane = tid & 31;
  const float inv_sqrt_r = 0.0625f;

  // q_feats (row-major into sF: needed as A for the out-GEMM and for d_tilde)
  {
    const int mt     = wave >> 1;
    const int ntbase = (wave & 1) * 8;
    const v16bf a0 = load_frag_a(sQ, Ec, mt * 16, 0);
    const v16bf a1 = load_frag_a(sQ, Ec, mt * 16, 32);
    const int tbase = mt * 16 + ((lane >> 4) << 3);
    const int ncol  = lane & 15;
#pragma unroll
    for (int nn = 0; nn < 8; ++nn) {
      const int nt = ntbase + nn;
      v8f c = {};
      c = wmma_bf16(a0, load_frag_b_t(sWt, Ec, 0,  nt * 16), c);
      c = wmma_bf16(a1, load_frag_b_t(sWt, Ec, 32, nt * 16), c);
      const int rcol = nt * 16 + ncol;
#pragma unroll
      for (int v = 0; v < 8; ++v) {
        const float f = __expf(c[v] - 0.5f * sNorm[tbase + v]) * inv_sqrt_r;
        sF[(tbase + v) * Rc + rcol] = (__bf16)f;
      }
    }
  }
  __syncthreads();

  // d_tilde per row (fp32, contiguous LDS row)
  if (tid < TILE) {
    float acc = 0.0f;
#pragma unroll 8
    for (int r = 0; r < Rc; ++r) acc += (float)sF[tid * Rc + r] * sKsum[r];
    sD[tid] = acc;
  }
  __syncthreads();

  // out_tile = q_feats @ kv : [TILE x Ec]. Wave owns (mt, nt-pair);
  // A fragment shared across both nt accumulators per k-step.
  {
    const int mt  = wave >> 1;
    const int nt0 = (wave & 1) * 2;
    v8f c0 = {}, c1 = {};
#pragma unroll
    for (int k0 = 0; k0 < Rc; k0 += 32) {
      const v16bf a = load_frag_a(sF, Rc, mt * 16, k0);
      c0 = wmma_bf16(a, load_frag_b_t(sKVt, Rc, k0, (nt0 + 0) * 16), c0);
      c1 = wmma_bf16(a, load_frag_b_t(sKVt, Rc, k0, (nt0 + 1) * 16), c1);
    }
    const int rowb = (lane >> 4) << 3;
    const int ncol = lane & 15;
#pragma unroll
    for (int v = 0; v < 8; ++v) {
      const int trow = mt * 16 + rowb + v;
      const float inv_d = 1.0f / sD[trow];
      out[((size_t)bh * Tc + t0 + trow) * Ec + (nt0 * 16 + ncol)]       = c0[v] * inv_d;
      out[((size_t)bh * Tc + t0 + trow) * Ec + ((nt0 + 1) * 16 + ncol)] = c1[v] * inv_d;
    }
  }
}

// ---------------- host launcher ----------------
extern "C" void kernel_launch(void* const* d_in, const int* in_sizes, int n_in,
                              void* d_out, int out_size, void* d_ws, size_t ws_size,
                              hipStream_t stream) {
  (void)in_sizes; (void)n_in; (void)out_size; (void)ws_size;
  const float* q = (const float*)d_in[0];
  const float* k = (const float*)d_in[1];
  const float* v = (const float*)d_in[2];
  const float* w = (const float*)d_in[3];
  float* out = (float*)d_out;

  float* gkv   = (float*)d_ws;                       // [B*H][R][E]
  float* gksum = gkv + (size_t)Bc * Hc * Rc * Ec;    // [B*H][R]
  const int nzero = Bc * Hc * Rc * Ec + Bc * Hc * Rc;

  zero_ws_kernel<<<(nzero + 255) / 256, 256, 0, stream>>>(gkv, nzero);

  constexpr size_t lds1 =
      (size_t)(Rc * Ec + TILE * Ec + Ec * TILE + Rc * TILE) * sizeof(__bf16) +
      (size_t)TILE * sizeof(float);                                   // 82176 B
  constexpr size_t lds2 =
      (size_t)(Rc * Ec + TILE * Ec + TILE * Rc + Ec * Rc) * sizeof(__bf16) +
      (size_t)(Rc + 2 * TILE) * sizeof(float);                        // 108032 B

  const dim3 grid(Bc * Hc * NT);  // 2048 blocks
  perf_pass1<<<grid, NTHREADS, lds1, stream>>>(k, v, w, gkv, gksum);
  perf_pass2<<<grid, NTHREADS, lds2, stream>>>(q, w, gkv, gksum, out);
}
